// WaveNet_24747601559557
// MI455X (gfx1250) — matrix operand order repaired and checked by reading, not compiled, not run on hardware
//
#include <hip/hip_runtime.h>

// ---------------------------------------------------------------------------
// WaveNet forward, fully fused in LDS, WMMA bf16 (gfx1250 / CDNA5, wave32).
//   grid:  B/2 blocks, 256 threads (8 waves) each
//   LDS :  h double-buffer [2][2*784][32] bf16 + per-wave gating scratch
// ---------------------------------------------------------------------------

typedef __attribute__((ext_vector_type(16))) __bf16 v16bf;
typedef __attribute__((ext_vector_type(8)))  __bf16 bh8;
typedef __attribute__((ext_vector_type(8)))  float  v8f;

#define SEQ        784
#define RCH        32
#define SPB        2            // samples per block
#define THREADS    256
#define WAVES      8
#define NTILES     ((SPB * SEQ) / 16)   // 98
#define TPS        (SEQ / 16)           // 49 time-tiles per sample

__device__ __constant__ int kDil[10] = {1, 2, 4, 8, 16, 1, 2, 4, 8, 16};

static __device__ __forceinline__ v16bf zero16() {
  v16bf r;
#pragma unroll
  for (int j = 0; j < 16; ++j) r[j] = (__bf16)0.0f;
  return r;
}

__global__ __launch_bounds__(THREADS)
void wavenet_fused_kernel(const float* __restrict__ x,
                          const float* __restrict__ w_causal,   // [32][1][2]
                          const float* __restrict__ b_causal,   // [32]
                          const float* __restrict__ w_dil,      // [10][64][32][2]
                          const float* __restrict__ b_dil,      // [10][64]
                          const float* __restrict__ w_res,      // [10][32][32]
                          const float* __restrict__ b_res,      // [10][32]
                          const float* __restrict__ w_out,      // [1][32]
                          const float* __restrict__ b_out,      // [1]
                          float* __restrict__ out)              // [B][784]
{
  extern __shared__ __align__(32) char smem_raw[];
  __bf16* hA   = (__bf16*)smem_raw;                 // [SPB*SEQ][32]
  __bf16* hB   = hA + SPB * SEQ * RCH;              // [SPB*SEQ][32]
  __bf16* gbuf = hB + SPB * SEQ * RCH;              // [WAVES][16][32]

  const int tid   = threadIdx.x;
  const int lane  = tid & 31;
  const int wave  = tid >> 5;
  const int n     = lane & 15;     // WMMA column (N) / B-matrix column
  const int hi    = lane >> 4;     // lane half
  const int khalf = hi * 8;        // C-layout row base for this lane half

  // ---------------- Phase 0: causal conv (1 -> 32 ch) into hA --------------
  for (int p = tid; p < SPB * SEQ; p += THREADS) {
    const int s  = p / SEQ;
    const int t  = p - s * SEQ;
    const int gs = blockIdx.x * SPB + s;
    const float x2 = (t >= 2) ? x[gs * SEQ + t - 2] : 0.0f;
    const float x1 = (t >= 1) ? x[gs * SEQ + t - 1] : 0.0f;
    __bf16* dst = hA + p * RCH;
#pragma unroll
    for (int c = 0; c < RCH; ++c)
      dst[c] = (__bf16)(w_causal[2 * c] * x2 + w_causal[2 * c + 1] * x1 + b_causal[c]);
  }
  __syncthreads();

  __bf16* hcur = hA;
  __bf16* hnxt = hB;
  __bf16* gme  = gbuf + wave * 16 * RCH;

  // ---------------- 10 dilated residual layers -----------------------------
  for (int L = 0; L < 10; ++L) {
    const int d = kDil[L];

    // --- dilated-conv A matrix: M=64 (4 tiles), K=64 = [h(t-d) | h(t)] -----
    // A 16-bit 16x32 layout: lane = M row; elems 0..15 -> K=hi*16..+15
    // (contiguous, matching B).
    v16bf Adil[4][2];
#pragma unroll
    for (int mt = 0; mt < 4; ++mt) {
      const int m = mt * 16 + n;
#pragma unroll
      for (int kc = 0; kc < 2; ++kc) {
        v16bf a;
#pragma unroll
        for (int idx = 0; idx < 16; ++idx) {
          const int kch = 2 * khalf + idx;  // 0..31 in chunk
          const int kg  = kc * 32 + kch;                      // 0..63
          const int c   = (kg < 32) ? kg : (kg - 32);
          const int tap = (kg < 32) ? 0 : 1;
          a[idx] = (__bf16)w_dil[((L * 64 + m) * 32 + c) * 2 + tap];
        }
        Adil[mt][kc] = a;
      }
    }
    // --- residual A matrix: M=32 (2 tiles), K=32 ---------------------------
    v16bf Ares[2];
#pragma unroll
    for (int mt = 0; mt < 2; ++mt) {
      const int m = mt * 16 + n;
      v16bf a;
#pragma unroll
      for (int idx = 0; idx < 16; ++idx) {
        const int k = 2 * khalf + idx;
        a[idx] = (__bf16)w_res[(L * 32 + m) * 32 + k];
      }
      Ares[mt] = a;
    }
    // --- biases in WMMA C layout (row M = mt*16 + khalf + r) ---------------
    v8f biasD[4], biasR[2];
#pragma unroll
    for (int mt = 0; mt < 4; ++mt)
#pragma unroll
      for (int r = 0; r < 8; ++r) biasD[mt][r] = b_dil[L * 64 + mt * 16 + khalf + r];
#pragma unroll
    for (int mt = 0; mt < 2; ++mt)
#pragma unroll
      for (int r = 0; r < 8; ++r) biasR[mt][r] = b_res[L * 32 + mt * 16 + khalf + r];

    // --- N-tile loop (wave-uniform trip count => EXEC all-ones at WMMA) ----
    for (int tile = wave; tile < NTILES; tile += WAVES) {
      const int s  = tile / TPS;
      const int tb = (tile - s * TPS) * 16;
      const int p  = tb + n;                                // column position
      const __bf16* hrow = hcur + (s * SEQ + p) * RCH;

      // B 16-bit 32x16 layout: lane = column; lane half hi holds 16
      // contiguous K (channels hi*16 .. hi*16+15) => one 32B LDS load.
      v16bf B0;  // K-chunk 0: shifted h(t-d), channels 0..31
      if (p >= d) {
        B0 = *(const v16bf*)(hcur + (s * SEQ + p - d) * RCH + hi * 16);
      } else {
        B0 = zero16();
      }
      const v16bf B1 = *(const v16bf*)(hrow + hi * 16);     // K-chunk 1: h(t)

      // z = Wdil * [h(t-d); h(t)] + b   (4 M-tiles x 2 K-chunks)
      v8f acc[4];
#pragma unroll
      for (int mt = 0; mt < 4; ++mt) {
        v8f c0 = __builtin_amdgcn_wmma_f32_16x16x32_bf16(
            false, Adil[mt][0], false, B0, (short)0, biasD[mt], false, false);
        acc[mt] = __builtin_amdgcn_wmma_f32_16x16x32_bf16(
            false, Adil[mt][1], false, B1, (short)0, c0, false, false);
      }

      // gated = tanh(z[0:32]) * sigmoid(z[32:64]); stash in B layout.
      // C layout: elem r of acc[mt] is channel mt*16 + khalf + r at column n.
#pragma unroll
      for (int g = 0; g < 2; ++g) {
        bh8 gv;
#pragma unroll
        for (int r = 0; r < 8; ++r) {
          const float th = tanhf(acc[g][r]);
          const float sg = 1.0f / (1.0f + __expf(-acc[g + 2][r]));
          gv[r] = (__bf16)(th * sg);
        }
        *(bh8*)(gme + n * RCH + g * 16 + khalf) = gv;
      }
      asm volatile("s_wait_dscnt 0" ::: "memory");   // cross-lane LDS handoff

      const v16bf Bg = *(const v16bf*)(gme + n * RCH + hi * 16);

      // h_next = h_cur + Wres * gated + b_res   (2 M-tiles)
#pragma unroll
      for (int mt = 0; mt < 2; ++mt) {
        v8f rc = __builtin_amdgcn_wmma_f32_16x16x32_bf16(
            false, Ares[mt], false, Bg, (short)0, biasR[mt], false, false);
        const bh8 old = *(const bh8*)(hrow + mt * 16 + khalf);
        bh8 upd;
#pragma unroll
        for (int r = 0; r < 8; ++r) upd[r] = (__bf16)((float)old[r] + rc[r]);
        *(bh8*)(hnxt + (s * SEQ + p) * RCH + mt * 16 + khalf) = upd;
      }
    }
    __syncthreads();
    __bf16* tmp = hcur; hcur = hnxt; hnxt = tmp;
  }

  // ---------------- Output projection (32 -> 1) ----------------------------
  for (int p = tid; p < SPB * SEQ; p += THREADS) {
    const int s  = p / SEQ;
    const int t  = p - s * SEQ;
    const int gs = blockIdx.x * SPB + s;
    const __bf16* hr = hcur + p * RCH;
    float acc = b_out[0];
#pragma unroll
    for (int c = 0; c < RCH; ++c) acc += w_out[c] * (float)hr[c];
    out[gs * SEQ + t] = acc;
  }
}

// ---------------------------------------------------------------------------
extern "C" void kernel_launch(void* const* d_in, const int* in_sizes, int n_in,
                              void* d_out, int out_size, void* d_ws, size_t ws_size,
                              hipStream_t stream) {
  const float* x        = (const float*)d_in[0];
  const float* w_causal = (const float*)d_in[1];
  const float* b_causal = (const float*)d_in[2];
  const float* w_dil    = (const float*)d_in[3];
  const float* b_dil    = (const float*)d_in[4];
  const float* w_res    = (const float*)d_in[5];
  const float* b_res    = (const float*)d_in[6];
  const float* w_out    = (const float*)d_in[7];
  const float* b_out    = (const float*)d_in[8];
  float* out = (float*)d_out;

  const int B      = in_sizes[0] / SEQ;          // 2048
  const int blocks = B / SPB;                    // 1024

  const size_t smem =
      (size_t)(2 * SPB * SEQ * RCH + WAVES * 16 * RCH) * sizeof(__bf16); // ~204 KB

  hipFuncSetAttribute((const void*)wavenet_fused_kernel,
                      hipFuncAttributeMaxDynamicSharedMemorySize, (int)smem);

  wavenet_fused_kernel<<<blocks, THREADS, smem, stream>>>(
      x, w_causal, b_causal, w_dil, b_dil, w_res, b_res, w_out, b_out, out);
}
